// GConvGRUCell_13185549599087
// MI455X (gfx1250) — matrix operand order repaired and checked by reading, not compiled, not run on hardware
//
#include <hip/hip_runtime.h>
#include <cstdint>
#include <cstddef>

#define BB 4
#define NN 20000
#define FF 128
#define HH 128
#define EE 320000
#define KK 256
#define MM (BB*NN)   // 80000 rows, divisible by 64

typedef __attribute__((ext_vector_type(16))) __bf16    v16bf;
typedef __attribute__((ext_vector_type(8)))  float     v8f;
typedef __attribute__((ext_vector_type(4)))  unsigned  u32x4;

union V16U { u32x4 q[2]; v16bf v; };

static __device__ __forceinline__ __bf16 f2bf(float f) { return (__bf16)f; }
static __device__ __forceinline__ float sigm(float x) { return 1.0f / (1.0f + __expf(-x)); }

// ---------------- degree / norm ----------------
__global__ __launch_bounds__(256) void k_deg_init(float* deg) {
  int i = blockIdx.x * 256 + threadIdx.x;
  if (i < NN) deg[i] = 1.0f;                       // self-loop contribution
}
__global__ __launch_bounds__(256) void k_deg_edges(const int* __restrict__ ei, float* deg) {
  int e = blockIdx.x * 256 + threadIdx.x;
  if (e < EE) atomicAdd(&deg[ei[EE + e]], 1.0f);   // dst row of edge_index
}
__global__ __launch_bounds__(256) void k_deg_fin(float* deg) {
  int i = blockIdx.x * 256 + threadIdx.x;
  if (i < NN) deg[i] = rsqrtf(deg[i]);             // deg >= 1 always (self-loops)
}

// ---------------- bf16 conversion ----------------
// Wzrt: [256 cols][256 K] = transpose of [Wz | Wr]; Wht: [128 cols][256 K] = Wh^T
__global__ __launch_bounds__(256) void k_cvt_weights(const float* __restrict__ Wz,
                                                     const float* __restrict__ Wr,
                                                     const float* __restrict__ Wh,
                                                     __bf16* __restrict__ Wzrt,
                                                     __bf16* __restrict__ Wht) {
  int i = blockIdx.x * 256 + threadIdx.x;
  if (i < 256 * 256) {
    int n = i >> 8, k = i & 255;
    Wzrt[i] = f2bf(n < 128 ? Wz[k * 128 + n] : Wr[k * 128 + (n - 128)]);
  } else if (i < 256 * 256 + 128 * 256) {
    int j = i - 256 * 256;
    int n = j >> 8, k = j & 255;
    Wht[j] = f2bf(Wh[k * 128 + n]);
  }
}

// xhb[row][0:128] = x, xhb[row][128:256] = h   (row = b*N + node)
__global__ __launch_bounds__(256) void k_cvt_xh(const float* __restrict__ x,
                                                const float* __restrict__ h,
                                                __bf16* __restrict__ xhb) {
  size_t i = (size_t)blockIdx.x * 256 + threadIdx.x;
  if (i >= (size_t)MM * KK) return;
  int c = (int)(i & 255);
  size_t row = i >> 8;
  xhb[i] = f2bf(c < 128 ? x[row * 128 + c] : h[row * 128 + (c - 128)]);
}

// ---------------- WMMA bf16 GEMM:  C[M x Nout] = A[M x 256] * Bt[Nout x 256]^T ----------------
// block = 8 waves; block tile = 64 rows x 128 cols; wave tile = 16 rows x 64 cols (4 WMMA n-tiles)
__global__ __launch_bounds__(256) void k_gemm(const __bf16* __restrict__ A,
                                              const __bf16* __restrict__ Bt,
                                              float* __restrict__ C, int Nout) {
  int lane = threadIdx.x & 31;
  int wave = threadIdx.x >> 5;
  int m0 = blockIdx.x * 64 + (wave & 3) * 16;
  int n0 = blockIdx.y * 128 + (wave >> 2) * 64;

  // A layout (16-bit A 16x32): lane<16 holds K {0..7,16..23}; lane>=16 holds K {8..15,24..31}
  int row   = m0 + (lane & 15);
  int aoff  = row * KK + ((lane & 16) ? 8 : 0);
  // B layout (32x16): lane<16 holds K {0..7,16..23}, lane>=16 holds K {8..15,24..31}, column = lane&15
  int ncol  = lane & 15;
  int koffb = (lane & 16) ? 8 : 0;

  v8f acc[4] = {};
  for (int k0 = 0; k0 < KK; k0 += 32) {
    V16U a;
    a.q[0] = *(const u32x4*)(A + aoff + k0);        // K = koff + 0..7
    a.q[1] = *(const u32x4*)(A + aoff + k0 + 16);   // K = koff + 16..23
#pragma unroll
    for (int t = 0; t < 4; t++) {
      const __bf16* bp = Bt + (size_t)(n0 + t * 16 + ncol) * KK + k0 + koffb;
      V16U b;
      b.q[0] = *(const u32x4*)bp;
      b.q[1] = *(const u32x4*)(bp + 16);
      acc[t] = __builtin_amdgcn_wmma_f32_16x16x32_bf16(
          false, a.v, false, b.v, (short)0, acc[t], false, false);
    }
  }

  int mb = m0 + ((lane & 16) ? 8 : 0);
#pragma unroll
  for (int t = 0; t < 4; t++) {
    int n = n0 + t * 16 + ncol;
#pragma unroll
    for (int i = 0; i < 8; i++) C[(size_t)(mb + i) * Nout + n] = acc[t][i];
  }
}

// ---------------- aggregation ----------------
// agg = t * dinv[node]^2 + bias  (self-loop term + bias), cshift = log2(Cdim)
__global__ __launch_bounds__(256) void k_agg_init(const float* __restrict__ t,
                                                  const float* __restrict__ dinv,
                                                  const float* __restrict__ b0,
                                                  const float* __restrict__ b1,
                                                  float* __restrict__ agg, int cshift) {
  size_t i = (size_t)blockIdx.x * 256 + threadIdx.x;
  int Cdim = 1 << cshift;
  if (i >= (size_t)MM * Cdim) return;
  int c = (int)(i & (Cdim - 1));
  size_t row = i >> cshift;
  int node = (int)(row % NN);
  float d = dinv[node];
  float bias = (c < 128) ? b0[c] : b1[c - 128];
  agg[i] = t[i] * d * d + bias;
}

// one wave per (edge, batch): gather t[src], scale by norm, atomic-add into agg[dst]
__global__ __launch_bounds__(256) void k_agg_edges(const int* __restrict__ ei,
                                                   const float* __restrict__ dinv,
                                                   const float* __restrict__ t,
                                                   float* __restrict__ agg, int Cdim) {
  unsigned gw = (blockIdx.x * 256u + threadIdx.x) >> 5;
  int lane = threadIdx.x & 31;
  if (gw >= (unsigned)EE * BB) return;
  int e = gw >> 2;
  int b = gw & 3;
  int s = ei[e];
  int d = ei[EE + e];
  float nrm = dinv[s] * dinv[d];
  const float4* ts = (const float4*)(t + ((size_t)b * NN + s) * Cdim);
  float* ad = agg + ((size_t)b * NN + d) * Cdim;
  int nv = Cdim >> 7;   // float4s per lane: 2 for C=256, 1 for C=128
  for (int v = 0; v < nv; v++) {
    float4 m = ts[lane + v * 32];
    int base = (lane + v * 32) * 4;
    atomicAdd(ad + base + 0, m.x * nrm);
    atomicAdd(ad + base + 1, m.y * nrm);
    atomicAdd(ad + base + 2, m.z * nrm);
    atomicAdd(ad + base + 3, m.w * nrm);
  }
}

// ---------------- elementwise ----------------
// z = sigmoid(agg_z) -> zout (d_out as temp); xhb second half = bf16(sigmoid(agg_r) * h)
__global__ __launch_bounds__(256) void k_zr_post(const float* __restrict__ agg,
                                                 const float* __restrict__ h,
                                                 float* __restrict__ zout,
                                                 __bf16* __restrict__ xhb) {
  size_t i = (size_t)blockIdx.x * 256 + threadIdx.x;
  if (i >= (size_t)MM * HH) return;
  int c = (int)(i & 127);
  size_t row = i >> 7;
  float z = sigm(agg[row * 256 + c]);
  float r = sigm(agg[row * 256 + 128 + c]);
  zout[i] = z;
  xhb[row * 256 + 128 + c] = f2bf(r * h[i]);
}

// out = (1-z)*h + z*tanh(agg_h), in-place over z buffer (d_out)
__global__ __launch_bounds__(256) void k_final(const float* __restrict__ aggh,
                                               const float* __restrict__ h,
                                               float* __restrict__ out) {
  size_t i = (size_t)blockIdx.x * 256 + threadIdx.x;
  if (i >= (size_t)MM * HH) return;
  float z = out[i];
  out[i] = (1.0f - z) * h[i] + z * tanhf(aggh[i]);
}

// ---------------- launch ----------------
extern "C" void kernel_launch(void* const* d_in, const int* in_sizes, int n_in,
                              void* d_out, int out_size, void* d_ws, size_t ws_size,
                              hipStream_t stream) {
  const float* x  = (const float*)d_in[0];
  const float* h  = (const float*)d_in[1];
  const float* Wz = (const float*)d_in[2];
  const float* bz = (const float*)d_in[3];
  const float* Wr = (const float*)d_in[4];
  const float* br = (const float*)d_in[5];
  const float* Wh = (const float*)d_in[6];
  const float* bh = (const float*)d_in[7];
  const int*   ei = (const int*)d_in[8];
  float* out = (float*)d_out;

  char* ws = (char*)d_ws;
  auto aln = [](size_t v) { return (v + 255) & ~(size_t)255; };
  size_t o = 0;
  float*  dinv  = (float*)(ws + o);  o = aln(o + (size_t)NN * 4);
  __bf16* xhb   = (__bf16*)(ws + o); o = aln(o + (size_t)MM * KK * 2);
  __bf16* Wzrt  = (__bf16*)(ws + o); o = aln(o + 256 * 256 * 2);
  __bf16* Wht   = (__bf16*)(ws + o); o = aln(o + 128 * 256 * 2);
  float*  tbuf  = (float*)(ws + o);  o = aln(o + (size_t)MM * 256 * 4);
  float*  aggzr = (float*)(ws + o);  o = aln(o + (size_t)MM * 256 * 4);
  float*  th    = tbuf;                         // reuse after tzr consumed
  float*  aggh  = tbuf + (size_t)MM * HH;       // second half of tbuf region

  // degree norm
  k_deg_init<<<(NN + 255) / 256, 256, 0, stream>>>(dinv);
  k_deg_edges<<<(EE + 255) / 256, 256, 0, stream>>>(ei, dinv);
  k_deg_fin<<<(NN + 255) / 256, 256, 0, stream>>>(dinv);

  // bf16 conversion
  k_cvt_weights<<<(256 * 256 + 128 * 256 + 255) / 256, 256, 0, stream>>>(Wz, Wr, Wh, Wzrt, Wht);
  k_cvt_xh<<<(int)(((size_t)MM * KK + 255) / 256), 256, 0, stream>>>(x, h, xhb);

  // fused z|r GEMM: [80000 x 256] @ [256 x 256]
  dim3 gzr(MM / 64, 2);
  k_gemm<<<gzr, 256, 0, stream>>>(xhb, Wzrt, tbuf, 256);

  // aggregate z|r
  k_agg_init<<<(int)(((size_t)MM * 256 + 255) / 256), 256, 0, stream>>>(tbuf, dinv, bz, br, aggzr, 8);
  int edge_blocks = (int)(((size_t)EE * BB * 32 + 255) / 256);
  k_agg_edges<<<edge_blocks, 256, 0, stream>>>(ei, dinv, tbuf, aggzr, 256);

  // z -> d_out (temp), r*h -> xhb second half (bf16)
  k_zr_post<<<(int)(((size_t)MM * HH + 255) / 256), 256, 0, stream>>>(aggzr, h, out, xhb);

  // h~ GEMM: [80000 x 256] @ [256 x 128]
  dim3 gh(MM / 64, 1);
  k_gemm<<<gh, 256, 0, stream>>>(xhb, Wht, th, 128);

  // aggregate h~
  k_agg_init<<<(int)(((size_t)MM * HH + 255) / 256), 256, 0, stream>>>(th, dinv, bh, bh, aggh, 7);
  k_agg_edges<<<edge_blocks, 256, 0, stream>>>(ei, dinv, th, aggh, 128);

  // GRU combine
  k_final<<<(int)(((size_t)MM * HH + 255) / 256), 256, 0, stream>>>(aggh, h, out);
}
